// UlyssesAttention_21157008900554
// MI455X (gfx1250) — matrix operand order repaired and checked by reading, not compiled, not run on hardware
//
#include <hip/hip_runtime.h>
#include <stdint.h>

// ---------------- problem constants ----------------
constexpr int D_MODEL = 1024;
constexpr int N_HEADS = 16;
constexpr int H_DIM   = 64;
constexpr int BATCH   = 2;
constexpr int SEQ     = 2048;
constexpr int M_TOT   = BATCH * SEQ;          // 4096

#define LOG2E    1.4426950408889634f
#define QK_SCALE (LOG2E / 8.0f)               // log2(e) / sqrt(64)

typedef __bf16 bf16_t;
typedef __attribute__((ext_vector_type(16))) __bf16 v16bf;
typedef __attribute__((ext_vector_type(8)))  float  v8f;

union Frag {
  v16bf v;
  uint4 q[2];
  unsigned u32[8];
  unsigned short h[16];
};

static __device__ __forceinline__ unsigned short f2bf(float f) {
  unsigned u = __float_as_uint(f);
  u += 0x7FFFu + ((u >> 16) & 1u);             // round-to-nearest-even
  return (unsigned short)(u >> 16);
}
static __device__ __forceinline__ unsigned pack2bf(float a, float b) {
  return (unsigned)f2bf(a) | ((unsigned)f2bf(b) << 16);
}
static __device__ __forceinline__ v8f vzero8() {
  v8f z;
#pragma unroll
  for (int i = 0; i < 8; ++i) z[i] = 0.0f;
  return z;
}
static __device__ __forceinline__ v8f wmma_bf16(const Frag& a, const Frag& b, v8f c) {
  // D = A(16x32 bf16) * B(32x16 bf16) + C(16x16 f32)
  return __builtin_amdgcn_wmma_f32_16x16x32_bf16(false, a.v, false, b.v,
                                                 (short)0, c, false, false);
}

// ---- CDNA5 async global->LDS copy (ASYNCcnt-tracked, no VGPR data) ----
static __device__ __forceinline__ unsigned lds_off(const void* p) {
  return (unsigned)(unsigned long long)(__attribute__((address_space(3))) const char*)p;
}
static __device__ __forceinline__ void async_load_b128(const void* gaddr, unsigned laddr) {
  asm volatile("global_load_async_to_lds_b128 %0, %1, off"
               :: "v"(laddr), "v"(gaddr) : "memory");
}
static __device__ __forceinline__ void wait_asynccnt0() {
  asm volatile("s_wait_asynccnt 0x0" ::: "memory");
}

// ---------------- fp32 -> bf16 converters ----------------
__global__ __launch_bounds__(256) void cvt_f32_bf16(const float* __restrict__ src,
                                                    bf16_t* __restrict__ dst) {
  int i = (blockIdx.x * 256 + threadIdx.x) * 8;
  float4 a = *(const float4*)(src + i);
  float4 b = *(const float4*)(src + i + 4);
  union { uint4 q; unsigned short h[8]; } o;
  o.h[0] = f2bf(a.x); o.h[1] = f2bf(a.y); o.h[2] = f2bf(a.z); o.h[3] = f2bf(a.w);
  o.h[4] = f2bf(b.x); o.h[5] = f2bf(b.y); o.h[6] = f2bf(b.z); o.h[7] = f2bf(b.w);
  *(uint4*)(dst + i) = o.q;
}

// Pack Wq|Wk|Wv into one [1024][3072] bf16 matrix
__global__ __launch_bounds__(256) void pack_wqkv(const float* __restrict__ Wq,
                                                 const float* __restrict__ Wk,
                                                 const float* __restrict__ Wv,
                                                 bf16_t* __restrict__ dst) {
  int i   = (blockIdx.x * 256 + threadIdx.x) * 8;   // index in [1024 x 3072]
  int k   = i / 3072;
  int col = i % 3072;
  int which = col >> 10;
  int c     = col & 1023;
  const float* W = (which == 0) ? Wq : (which == 1) ? Wk : Wv;
  const float* s = W + (size_t)k * 1024 + c;
  float4 a = *(const float4*)s;
  float4 b = *(const float4*)(s + 4);
  union { uint4 q; unsigned short h[8]; } o;
  o.h[0] = f2bf(a.x); o.h[1] = f2bf(a.y); o.h[2] = f2bf(a.z); o.h[3] = f2bf(a.w);
  o.h[4] = f2bf(b.x); o.h[5] = f2bf(b.y); o.h[6] = f2bf(b.z); o.h[7] = f2bf(b.w);
  *(uint4*)(dst + i) = o.q;
}

// ---------------- tiled WMMA GEMM ----------------
// C[M x N] = A[M x 1024] * Bw[1024 x N]  (bf16 in, f32 accum)
// A tile is staged with global_load_async_to_lds_b128 (ASYNCcnt path);
// W tile is transposed through VGPRs (async cannot transpose).
// MODE 0: fused QKV -> scatter bf16 into q/k/v [B,H,S,hd], fold bias + QK scale into Q
// MODE 1: out projection -> f32 output + bias
template <int MODE>
__global__ __launch_bounds__(256) void gemm_wmma(
    const bf16_t* __restrict__ A, const bf16_t* __restrict__ Bw, int N,
    const float* __restrict__ b0, const float* __restrict__ b1, const float* __restrict__ b2,
    bf16_t* __restrict__ oq, bf16_t* __restrict__ ok, bf16_t* __restrict__ ov,
    float* __restrict__ of) {
  __shared__ __align__(16) bf16_t As[128][40];   // [m][k] padded (80B rows)
  __shared__ __align__(16) bf16_t Bs[128][40];   // [n][k] transposed W tile

  const int tid  = threadIdx.x;
  const int lane = tid & 31;
  const int wave = tid >> 5;
  const int wm = wave & 1;          // 2 waves along M
  const int wn = wave >> 1;         // 4 waves along N
  const int l16 = lane & 15;
  const int lhi = lane >> 4;
  const int m0 = blockIdx.y * 128;
  const int n0 = blockIdx.x * 128;

  // per-thread staging coordinates (invariant across k-steps)
  const int arow = tid >> 1, aseg = tid & 1;
  const unsigned a_l0 = lds_off(&As[arow][aseg * 16]);
  const unsigned a_l1 = a_l0 + 16;

  v8f acc[4][2];
#pragma unroll
  for (int mi = 0; mi < 4; ++mi)
#pragma unroll
    for (int ni = 0; ni < 2; ++ni) acc[mi][ni] = vzero8();

  for (int k0 = 0; k0 < 1024; k0 += 32) {
    // --- stage A tile 128x32 via async global->LDS (2 x b128 per thread) ---
    {
      const bf16_t* g = A + (size_t)(m0 + arow) * 1024 + k0 + aseg * 16;
      async_load_b128(g, a_l0);
      async_load_b128(g + 8, a_l1);
    }
    // --- stage transposed W tile: Bs[n][k] = Bw[k0+k][n0+n] ---
    if (tid < 128) {
      int kk  = (tid >> 3) << 1;     // even k row
      int seg = tid & 7;
      const bf16_t* g0 = Bw + (size_t)(k0 + kk) * N + n0 + seg * 16;
      const bf16_t* g1 = g0 + N;
      union { uint4 q[2]; unsigned short h[16]; } r0, r1;
      r0.q[0] = *(const uint4*)g0; r0.q[1] = *(const uint4*)(g0 + 8);
      r1.q[0] = *(const uint4*)g1; r1.q[1] = *(const uint4*)(g1 + 8);
#pragma unroll
      for (int e = 0; e < 16; ++e)
        *(unsigned*)&Bs[seg * 16 + e][kk] =
            (unsigned)r0.h[e] | ((unsigned)r1.h[e] << 16);
    }
    wait_asynccnt0();
    __syncthreads();

    Frag af[4], bfg[2];
#pragma unroll
    for (int mi = 0; mi < 4; ++mi) {
      int row = wm * 64 + mi * 16 + l16;
      int kb  = lhi * 8;
      af[mi].q[0] = *(const uint4*)&As[row][kb];        // K = kb..kb+7
      af[mi].q[1] = *(const uint4*)&As[row][kb + 16];   // K = kb+16..kb+23
    }
#pragma unroll
    for (int ni = 0; ni < 2; ++ni) {
      int col = wn * 32 + ni * 16 + l16;
      int kb  = lhi * 8;
      bfg[ni].q[0] = *(const uint4*)&Bs[col][kb];       // K = kb..kb+7
      bfg[ni].q[1] = *(const uint4*)&Bs[col][kb + 16];  // K = kb+16..kb+23
    }
#pragma unroll
    for (int mi = 0; mi < 4; ++mi)
#pragma unroll
      for (int ni = 0; ni < 2; ++ni)
        acc[mi][ni] = wmma_bf16(af[mi], bfg[ni], acc[mi][ni]);
    __syncthreads();
  }

  // --- epilogue ---
#pragma unroll
  for (int mi = 0; mi < 4; ++mi) {
#pragma unroll
    for (int ni = 0; ni < 2; ++ni) {
      int nb = n0 + wn * 32 + ni * 16;
      if (MODE == 0) {
        int which = nb >> 10;
        int rem0  = nb & 1023;
        const float* bias = (which == 0) ? b0 : (which == 1) ? b1 : b2;
        bf16_t* outp      = (which == 0) ? oq : (which == 1) ? ok : ov;
        float sc          = (which == 0) ? QK_SCALE : 1.0f;
        unsigned short* op = (unsigned short*)outp;
#pragma unroll
        for (int r = 0; r < 8; ++r) {
          int m = m0 + wm * 64 + mi * 16 + r + (lhi ? 8 : 0);
          int n = rem0 + l16;
          float v = (acc[mi][ni][r] + bias[n]) * sc;
          int h = n >> 6, d = n & 63;
          int bb = m >> 11, s = m & 2047;
          op[(((size_t)(bb * N_HEADS + h)) * SEQ + s) * H_DIM + d] = f2bf(v);
        }
      } else {
#pragma unroll
        for (int r = 0; r < 8; ++r) {
          int m = m0 + wm * 64 + mi * 16 + r + (lhi ? 8 : 0);
          int n = nb + l16;
          of[(size_t)m * D_MODEL + n] = acc[mi][ni][r] + b0[n];
        }
      }
    }
  }
}

// ---------------- flash attention ----------------
// Per workgroup: 128 query rows of one (b,h); 8 waves x 16 rows each.
// K tile (128x64) staged ONCE per workgroup via async global->LDS (removes the
// 8x redundant per-wave global reads); V^T staged via VGPR transpose.
// Computes S^T = K * Q^T, base-2 online softmax, O^T = V^T * P^T with an
// in-register C-layout -> B-fragment transpose of P.
__global__ __launch_bounds__(256) void flash_attn(
    const bf16_t* __restrict__ Q, const bf16_t* __restrict__ K,
    const bf16_t* __restrict__ V, const float* __restrict__ mask,
    bf16_t* __restrict__ O /* attn bf16 [B,S,D] */) {
  __shared__ __align__(16) bf16_t Ks[128][72];   // [k][d] K tile (144B rows)
  __shared__ __align__(16) bf16_t Vt[64][136];   // [d][k] transposed V tile

  const int tid  = threadIdx.x;
  const int lane = tid & 31;
  const int wave = tid >> 5;
  const int l16  = lane & 15;
  const int lhi  = lane >> 4;
  const int bh = blockIdx.y;
  const int b  = bh >> 4;
  const int h  = bh & 15;
  const int qb = blockIdx.x * 128 + wave * 16;   // this wave's 16 query rows

  const bf16_t* qp = Q + (size_t)bh * SEQ * H_DIM;
  const bf16_t* kp = K + (size_t)bh * SEQ * H_DIM;
  const bf16_t* vp = V + (size_t)bh * SEQ * H_DIM;
  const float*  mp = mask + (size_t)bh * SEQ * SEQ + (size_t)(qb + l16) * SEQ;

  // Q^T B-fragments (persist): lane = column q, d in table order (8h+s | 16+8h+s-8)
  Frag qf[2];
#pragma unroll
  for (int c = 0; c < 2; ++c) {
    const bf16_t* g = qp + (size_t)(qb + l16) * H_DIM + c * 32 + lhi * 8;
    qf[c].q[0] = *(const uint4*)g;
    qf[c].q[1] = *(const uint4*)(g + 16);
  }

  v8f o[4];
#pragma unroll
  for (int df = 0; df < 4; ++df) o[df] = vzero8();
  float m_run = -1e30f, l_run = 0.0f;

  for (int kt = 0; kt < SEQ; kt += 128) {
    // --- stage K tile 128x64 via async global->LDS (4 x b128 per thread) ---
#pragma unroll
    for (int t = 0; t < 4; ++t) {
      int chunk = tid + t * 256;          // 1024 chunks of 8 bf16
      int row = chunk >> 3, seg = chunk & 7;
      async_load_b128(kp + (size_t)(kt + row) * H_DIM + seg * 8,
                      lds_off(&Ks[row][seg * 8]));
    }
    // --- stage V^T tile: Vt[d][k] = V[kt+k][d] ---
    {
      int kr  = (tid >> 2) << 1;   // even k row 0..126
      int seg = tid & 3;
      const bf16_t* g0 = vp + (size_t)(kt + kr) * H_DIM + seg * 16;
      const bf16_t* g1 = g0 + H_DIM;
      union { uint4 q[2]; unsigned short h[16]; } r0, r1;
      r0.q[0] = *(const uint4*)g0; r0.q[1] = *(const uint4*)(g0 + 8);
      r1.q[0] = *(const uint4*)g1; r1.q[1] = *(const uint4*)(g1 + 8);
#pragma unroll
      for (int e = 0; e < 16; ++e)
        *(unsigned*)&Vt[seg * 16 + e][kr] =
            (unsigned)r0.h[e] | ((unsigned)r1.h[e] << 16);
    }
    wait_asynccnt0();
    __syncthreads();

    // --- scores: S^T tile 128(k) x 16(q), 8 C-fragments ---
    v8f s[8];
#pragma unroll
    for (int kr = 0; kr < 8; ++kr) {
      v8f c = vzero8();
#pragma unroll
      for (int cc = 0; cc < 2; ++cc) {
        Frag ka;   // A = K rows (table-order d) from LDS
        const bf16_t* ls = &Ks[kr * 16 + l16][cc * 32 + lhi * 8];
        ka.q[0] = *(const uint4*)ls;
        ka.q[1] = *(const uint4*)(ls + 16);
        c = wmma_bf16(ka, qf[cc], c);
      }
      // + mask * log2(e)  (softmax runs in base 2)
      const float* mrow = mp + kt + kr * 16 + (lhi ? 8 : 0);
      float4 ma = *(const float4*)mrow;
      float4 mb = *(const float4*)(mrow + 4);
      c[0] += ma.x * LOG2E; c[1] += ma.y * LOG2E;
      c[2] += ma.z * LOG2E; c[3] += ma.w * LOG2E;
      c[4] += mb.x * LOG2E; c[5] += mb.y * LOG2E;
      c[6] += mb.z * LOG2E; c[7] += mb.w * LOG2E;
      s[kr] = c;
    }

    // --- online softmax (k lives in registers + other lane half) ---
    float mx = s[0][0];
#pragma unroll
    for (int kr = 0; kr < 8; ++kr)
#pragma unroll
      for (int r = 0; r < 8; ++r) mx = fmaxf(mx, s[kr][r]);
    mx = fmaxf(mx, __shfl_xor(mx, 16, 32));
    float m_new = fmaxf(m_run, mx);
    float alpha = exp2f(m_run - m_new);
    float rs = 0.0f;
#pragma unroll
    for (int kr = 0; kr < 8; ++kr)
#pragma unroll
      for (int r = 0; r < 8; ++r) {
        float p = exp2f(s[kr][r] - m_new);
        s[kr][r] = p;
        rs += p;
      }
    rs += __shfl_xor(rs, 16, 32);
    l_run = l_run * alpha + rs;
    m_run = m_new;
#pragma unroll
    for (int df = 0; df < 4; ++df)
#pragma unroll
      for (int r = 0; r < 8; ++r) o[df][r] *= alpha;

    // --- O^T += V^T * P^T : build P B-fragments in registers ---
#pragma unroll
    for (int kc = 0; kc < 4; ++kc) {
      Frag pf;
      v8f c0 = s[2 * kc], c1 = s[2 * kc + 1];
#pragma unroll
      for (int j = 0; j < 4; ++j) {
        unsigned lo  = pack2bf(c0[2 * j], c0[2 * j + 1]);
        unsigned hi  = pack2bf(c1[2 * j], c1[2 * j + 1]);
        unsigned xlo = (unsigned)__shfl_xor((int)lo, 16, 32);
        unsigned xhi = (unsigned)__shfl_xor((int)hi, 16, 32);
        pf.u32[j]     = lhi ? xhi : lo;   // k = 2j,2j+1 | 16+2j,17+2j
        pf.u32[j + 4] = lhi ? hi  : xlo;  // k = 8+2j,9+2j | 24+2j,25+2j
      }
#pragma unroll
      for (int df = 0; df < 4; ++df) {
        Frag va;   // A = V^T rows (d), contiguous k from LDS
        const bf16_t* ls = &Vt[df * 16 + l16][kc * 32 + lhi * 16];
        va.q[0] = *(const uint4*)ls;
        va.q[1] = *(const uint4*)(ls + 8);
        o[df] = wmma_bf16(va, pf, o[df]);
      }
    }
    __syncthreads();
  }

  // --- finalize: O = O^T / l, scatter bf16 into attn [B,S,H*hd] ---
  float inv = 1.0f / l_run;
  unsigned short* op = (unsigned short*)O;
  const int q = qb + l16;
#pragma unroll
  for (int df = 0; df < 4; ++df)
#pragma unroll
    for (int r = 0; r < 8; ++r) {
      int d = df * 16 + r + (lhi ? 8 : 0);
      op[(size_t)(b * SEQ + q) * D_MODEL + h * H_DIM + d] = f2bf(o[df][r] * inv);
    }
}

// ---------------- launcher ----------------
extern "C" void kernel_launch(void* const* d_in, const int* in_sizes, int n_in,
                              void* d_out, int out_size, void* d_ws, size_t ws_size,
                              hipStream_t stream) {
  (void)in_sizes; (void)n_in; (void)out_size; (void)ws_size;
  const float* x    = (const float*)d_in[0];
  const float* mask = (const float*)d_in[1];
  const float* Wq = (const float*)d_in[2]; const float* bq = (const float*)d_in[3];
  const float* Wk = (const float*)d_in[4]; const float* bk = (const float*)d_in[5];
  const float* Wv = (const float*)d_in[6]; const float* bv = (const float*)d_in[7];
  const float* Wo = (const float*)d_in[8]; const float* bo = (const float*)d_in[9];
  float* out = (float*)d_out;

  char* ws = (char*)d_ws;
  const size_t MB = 1024 * 1024;
  bf16_t* x_bf    = (bf16_t*)(ws);              //  8 MB  [4096 x 1024]
  bf16_t* wqkv_bf = (bf16_t*)(ws + 8 * MB);     //  6 MB  [1024 x 3072]
  bf16_t* wo_bf   = (bf16_t*)(ws + 14 * MB);    //  2 MB  [1024 x 1024]
  bf16_t* q_bf    = (bf16_t*)(ws + 16 * MB);    //  8 MB  [B,H,S,hd]
  bf16_t* k_bf    = (bf16_t*)(ws + 24 * MB);    //  8 MB
  bf16_t* v_bf    = (bf16_t*)(ws + 32 * MB);    //  8 MB
  bf16_t* attn_bf = (bf16_t*)(ws + 40 * MB);    //  8 MB  [4096 x 1024]

  // 1. convert inputs to bf16
  cvt_f32_bf16<<<(M_TOT * D_MODEL) / (8 * 256), 256, 0, stream>>>(x, x_bf);
  pack_wqkv<<<(1024 * 3072) / (8 * 256), 256, 0, stream>>>(Wq, Wk, Wv, wqkv_bf);
  cvt_f32_bf16<<<(1024 * 1024) / (8 * 256), 256, 0, stream>>>(Wo, wo_bf);

  // 2. fused QKV projection (M=4096, N=3072)
  gemm_wmma<0><<<dim3(3072 / 128, M_TOT / 128), 256, 0, stream>>>(
      x_bf, wqkv_bf, 3072, bq, bk, bv, q_bf, k_bf, v_bf, nullptr);

  // 3. flash attention
  flash_attn<<<dim3(SEQ / 128, BATCH * N_HEADS), 256, 0, stream>>>(
      q_bf, k_bf, v_bf, mask, attn_bf);

  // 4. output projection (M=4096, N=1024) -> f32 out + bias
  gemm_wmma<1><<<dim3(D_MODEL / 128, M_TOT / 128), 256, 0, stream>>>(
      attn_bf, wo_bf, 1024, bo, nullptr, nullptr, nullptr, nullptr, nullptr, out);
}
